// AWQW4A16Linear_20246475833568
// MI455X (gfx1250) — matrix operand order repaired and checked by reading, not compiled, not run on hardware
//
#include <hip/hip_runtime.h>
#include <hip/hip_bf16.h>

// ---------------- shared types / helpers ----------------
typedef __attribute__((ext_vector_type(16))) __bf16 bf16x16;
typedef __attribute__((ext_vector_type(8)))  __bf16 bf16x8;
typedef __attribute__((ext_vector_type(4)))  __bf16 bf16x4;
typedef __attribute__((ext_vector_type(2)))  __bf16 bf16x2;
typedef __attribute__((ext_vector_type(8)))  float  f32x8;
typedef __attribute__((ext_vector_type(4)))  int    i32x4v;

union FragAB { bf16x16 v; bf16x8 h[2]; };
union BF4    { bf16x4  v; bf16x2 h[2]; };
union BF8    { bf16x8  v; bf16x2 q[4]; };

#define NTHREADS 256

__device__ __forceinline__ __bf16 f2bf(float f) {
    union { float f; unsigned u; } a; a.f = f;
    unsigned r = a.u + 0x7FFFu + ((a.u >> 16) & 1u);   // round-to-nearest-even
    union { unsigned short s; __bf16 b; } o;
    o.s = (unsigned short)(r >> 16);
    return o.b;
}

__device__ __forceinline__ bf16x2 pk_bf16(float lo, float hi) {
#if __has_builtin(__builtin_amdgcn_cvt_pk_bf16_f32)
    return __builtin_amdgcn_cvt_pk_bf16_f32(lo, hi);
#else
    bf16x2 r; r[0] = f2bf(lo); r[1] = f2bf(hi); return r;
#endif
}

// ---------------- CDNA5 async global->LDS copy (16B per lane) ----------------
#if __has_builtin(__builtin_amdgcn_global_load_async_to_lds_b128)
#define ASYNC_OK 1
#endif

typedef __attribute__((address_space(1))) i32x4v glob_i32x4;   // prints as "__device__"
typedef __attribute__((address_space(3))) i32x4v lds_i32x4;    // prints as "__shared__"

__device__ __forceinline__ void async_cp16(const void* g, void* l) {
#ifdef ASYNC_OK
    __builtin_amdgcn_global_load_async_to_lds_b128(
        (glob_i32x4*)g, (lds_i32x4*)l, 0, 0);
#else
    *(bf16x8*)l = *(const bf16x8*)g;          // sync fallback via VGPRs
#endif
}

__device__ __forceinline__ void wait_async() {
#ifdef ASYNC_OK
 #if __has_builtin(__builtin_amdgcn_s_wait_asynccnt)
    __builtin_amdgcn_s_wait_asynccnt(0);
 #else
    asm volatile("s_wait_asynccnt 0" ::: "memory");
 #endif
#endif
}

// =====================================================================
// Pre-pass 1: x fp32 -> bf16 (done once; x is re-read 96x by GEMM blocks)
// =====================================================================
__global__ __launch_bounds__(NTHREADS)
void convert_x_kernel(const float* __restrict__ x, __bf16* __restrict__ xbf, long total) {
    long i = ((long)blockIdx.x * NTHREADS + threadIdx.x) * 8;
    if (i >= total) return;
    const float4 a = *(const float4*)(x + i);
    const float4 b = *(const float4*)(x + i + 4);
    BF8 v;
    v.q[0] = pk_bf16(a.x, a.y); v.q[1] = pk_bf16(a.z, a.w);
    v.q[2] = pk_bf16(b.x, b.y); v.q[3] = pk_bf16(b.z, b.w);
    *(bf16x8*)(xbf + i) = v.v;
}

// =====================================================================
// Pre-pass 2: int4 AWQ -> bf16 [N,K]. qweight stores one byte per int32,
// so the bf16 output is the SAME size as the input: traffic-neutral, and
// it removes the 32x-redundant per-block dequant VALU from the GEMM loop.
// =====================================================================
__global__ __launch_bounds__(NTHREADS)
void dequant_w_kernel(const int* __restrict__ qw, const float* __restrict__ wscales,
                      const float* __restrict__ wzeros, __bf16* __restrict__ wbf,
                      int N, int K) {
    const int Kh  = K >> 1;            // packed bytes per row
    const int Jq  = Kh >> 2;           // int4 chunks per row
    long t = (long)blockIdx.x * NTHREADS + threadIdx.x;
    if (t >= (long)N * Jq) return;
    const int n  = (int)(t / Jq);
    const int j0 = (int)(t % Jq) * 4;          // first packed byte index
    const int g  = j0 >> 6;                    // group = (2*j0)/128
    const float s = wscales[(size_t)g * N + n];
    const float c = -wzeros[(size_t)g * N + n] * s;    // w = nib*s + c
    const int4 q4 = *(const int4*)(qw + (size_t)n * Kh + j0);
    BF8 v;
    int q;
    q = q4.x; v.q[0] = pk_bf16(__builtin_fmaf((float)( q       & 0xF), s, c),
                               __builtin_fmaf((float)((q >> 4) & 0xF), s, c));
    q = q4.y; v.q[1] = pk_bf16(__builtin_fmaf((float)( q       & 0xF), s, c),
                               __builtin_fmaf((float)((q >> 4) & 0xF), s, c));
    q = q4.z; v.q[2] = pk_bf16(__builtin_fmaf((float)( q       & 0xF), s, c),
                               __builtin_fmaf((float)((q >> 4) & 0xF), s, c));
    q = q4.w; v.q[3] = pk_bf16(__builtin_fmaf((float)( q       & 0xF), s, c),
                               __builtin_fmaf((float)((q >> 4) & 0xF), s, c));
    *(bf16x8*)(wbf + (size_t)n * K + 2 * j0) = v.v;
}

// =====================================================================
// Main GEMM: pure bf16, 256x128x128 tiles, async global->LDS double buffer
// =====================================================================
#define GBM 256
#define GBN 128
#define GBK 128
#define GLDA 136       // padded LDS row stride (bf16 elems); 272B = 16B-aligned

__global__ __launch_bounds__(NTHREADS)
void gemm_bf16_kernel(const __bf16* __restrict__ xbf,   // [M,K]
                      const __bf16* __restrict__ wbf,   // [N,K]
                      const float*  __restrict__ bias,  // [N]
                      float*        __restrict__ out,   // [M,N]
                      int M, int N, int K)
{
    extern __shared__ __bf16 smem[];
    __bf16* sA = smem;                          // [2][GBM*GLDA]
    __bf16* sW = smem + 2 * (GBM * GLDA);       // [2][GBN*GLDA]

    const int tid   = threadIdx.x;
    const int lane  = tid & 31;
    const int wave  = tid >> 5;
    const int wm    = wave & 3;                 // 4 M slots x 64 rows
    const int wn    = wave >> 2;                // 2 N slots x 64 cols
    const int l16   = lane & 15;
    const int lhalf = lane >> 4;

    const int bm = blockIdx.y * GBM;
    const int bn = blockIdx.x * GBN;
    const int KT = K / GBK;

    auto issue_tile = [&](int kt, int buf) {
        const __bf16* xg = xbf + (size_t)bm * K + (size_t)kt * GBK;
        __bf16* la = sA + buf * (GBM * GLDA);
#pragma unroll
        for (int p = 0; p < 16; ++p) {          // 256x128 bf16 = 4096 x 16B chunks
            int idx = p * NTHREADS + tid;
            int r = idx >> 4, c = (idx & 15) << 3;
            async_cp16(xg + (size_t)r * K + c, la + r * GLDA + c);
        }
        const __bf16* wg = wbf + (size_t)bn * K + (size_t)kt * GBK;
        __bf16* lw = sW + buf * (GBN * GLDA);
#pragma unroll
        for (int p = 0; p < 8; ++p) {           // 128x128 bf16 = 2048 x 16B chunks
            int idx = p * NTHREADS + tid;
            int r = idx >> 4, c = (idx & 15) << 3;
            async_cp16(wg + (size_t)r * K + c, lw + r * GLDA + c);
        }
    };

    f32x8 acc[4][4];
    const f32x8 zero8 = {0.f, 0.f, 0.f, 0.f, 0.f, 0.f, 0.f, 0.f};
#pragma unroll
    for (int i = 0; i < 4; ++i)
#pragma unroll
        for (int j = 0; j < 4; ++j) acc[i][j] = zero8;

    issue_tile(0, 0);

    for (int kt = 0; kt < KT; ++kt) {
        wait_async();                     // tile kt landed in LDS (ASYNCcnt -> 0)
        __syncthreads();
        const int cur = kt & 1;

        if (kt + 1 < KT) issue_tile(kt + 1, cur ^ 1);   // overlap with WMMA below

        if (kt + 2 < KT) {                               // warm L2 for kt+2
            __builtin_prefetch(xbf + (size_t)(bm + tid) * K + (size_t)(kt + 2) * GBK, 0, 0);
            if (tid < GBN)
                __builtin_prefetch(wbf + (size_t)(bn + tid) * K + (size_t)(kt + 2) * GBK, 0, 0);
        }

        const __bf16* a = sA + cur * (GBM * GLDA);
        const __bf16* w = sW + cur * (GBN * GLDA);

#pragma unroll
        for (int ks = 0; ks < 4; ++ks) {
            const int kk = ks * 32;
            FragAB fa[4], fb[4];
            // A 16x32 bf16: lane-half 0 -> K[0..7],[16..23]; half 1 -> K[8..15],[24..31]
#pragma unroll
            for (int i = 0; i < 4; ++i) {
                const __bf16* base = a + (wm * 64 + i * 16 + l16) * GLDA + kk + lhalf * 8;
                fa[i].h[0] = *(const bf16x8*)(base);
                fa[i].h[1] = *(const bf16x8*)(base + 16);
            }
            // B 32x16 bf16: lane = column, lane-half 0 -> K[0..7],[16..23]; half 1 -> K[8..15],[24..31]
#pragma unroll
            for (int j = 0; j < 4; ++j) {
                const __bf16* base = w + (wn * 64 + j * 16 + l16) * GLDA + kk + lhalf * 8;
                fb[j].h[0] = *(const bf16x8*)(base);
                fb[j].h[1] = *(const bf16x8*)(base + 16);
            }
#pragma unroll
            for (int i = 0; i < 4; ++i)
#pragma unroll
                for (int j = 0; j < 4; ++j)
                    acc[i][j] = __builtin_amdgcn_wmma_f32_16x16x32_bf16(
                        false, fa[i].v, false, fb[j].v,
                        (short)0, acc[i][j], false, false);
        }
    }

    // Epilogue: NT stores keep xbf+wbf (134MB) resident in the 192MB L2
#pragma unroll
    for (int j = 0; j < 4; ++j) {
        const int n  = bn + wn * 64 + j * 16 + l16;
        const float b = bias[n];
#pragma unroll
        for (int i = 0; i < 4; ++i) {
            const int mrow = bm + wm * 64 + i * 16 + lhalf * 8;
#pragma unroll
            for (int r = 0; r < 8; ++r)
                __builtin_nontemporal_store(acc[i][j][r] + b,
                                            &out[(size_t)(mrow + r) * N + n]);
        }
    }
}

// =====================================================================
// Fallback: fused kernel for small ws_size
// =====================================================================
#define BM 128
#define BN 128
#define BK 128
#define LDA 136

__global__ __launch_bounds__(NTHREADS)
void awq_w4a16_fused_kernel(const float* __restrict__ x, const int* __restrict__ qw,
                            const float* __restrict__ wscales, const float* __restrict__ wzeros,
                            const float* __restrict__ bias, float* __restrict__ out,
                            int M, int N, int K)
{
    extern __shared__ __bf16 smem[];
    __bf16* sA = smem;
    __bf16* sW = smem + 2 * (BM * LDA);

    const int tid   = threadIdx.x;
    const int lane  = tid & 31;
    const int wave  = tid >> 5;
    const int wm    = wave & 1;
    const int wn    = wave >> 1;
    const int l16   = lane & 15;
    const int lhalf = lane >> 4;

    const int bm = blockIdx.y * BM;
    const int bn = blockIdx.x * BN;
    const int KT = K / BK;
    const int Kh = K >> 1;

    const int axc = (tid & 31) * 4;
    const int ayr = tid >> 5;
    const int wxc = (tid & 15) * 4;
    const int wyr = tid >> 4;

    float4 xr[16];
    int4   qr[8];
    float  sr[8], zr[8];

    auto load_tile = [&](int kt) {
        const float* xp = x + (size_t)bm * K + (size_t)kt * BK;
#pragma unroll
        for (int p = 0; p < 16; ++p)
            xr[p] = *(const float4*)(xp + (size_t)(p * 8 + ayr) * K + axc);
        const int* qp = qw + (size_t)bn * Kh + (size_t)kt * (BK / 2);
#pragma unroll
        for (int p = 0; p < 8; ++p) {
            int n = p * 16 + wyr;
            qr[p] = *(const int4*)(qp + (size_t)n * Kh + wxc);
            sr[p] = wscales[(size_t)kt * N + bn + n];
            zr[p] = wzeros [(size_t)kt * N + bn + n];
        }
    };

    auto store_tile = [&](int buf) {
        __bf16* a = sA + buf * (BM * LDA);
#pragma unroll
        for (int p = 0; p < 16; ++p) {
            BF4 t;
            t.h[0] = pk_bf16(xr[p].x, xr[p].y);
            t.h[1] = pk_bf16(xr[p].z, xr[p].w);
            *(bf16x4*)(a + (p * 8 + ayr) * LDA + axc) = t.v;
        }
        __bf16* w = sW + buf * (BN * LDA);
#pragma unroll
        for (int p = 0; p < 8; ++p) {
            int n = p * 16 + wyr;
            const float s = sr[p];
            const float c = -zr[p] * s;
            BF8 v; int q;
            q = qr[p].x; v.q[0] = pk_bf16(__builtin_fmaf((float)( q       & 0xF), s, c),
                                          __builtin_fmaf((float)((q >> 4) & 0xF), s, c));
            q = qr[p].y; v.q[1] = pk_bf16(__builtin_fmaf((float)( q       & 0xF), s, c),
                                          __builtin_fmaf((float)((q >> 4) & 0xF), s, c));
            q = qr[p].z; v.q[2] = pk_bf16(__builtin_fmaf((float)( q       & 0xF), s, c),
                                          __builtin_fmaf((float)((q >> 4) & 0xF), s, c));
            q = qr[p].w; v.q[3] = pk_bf16(__builtin_fmaf((float)( q       & 0xF), s, c),
                                          __builtin_fmaf((float)((q >> 4) & 0xF), s, c));
            *(bf16x8*)(w + n * LDA + wxc * 2) = v.v;
        }
    };

    f32x8 acc[4][2];
    const f32x8 zero8 = {0.f, 0.f, 0.f, 0.f, 0.f, 0.f, 0.f, 0.f};
#pragma unroll
    for (int i = 0; i < 4; ++i)
#pragma unroll
        for (int j = 0; j < 2; ++j) acc[i][j] = zero8;

    load_tile(0);
    store_tile(0);

    for (int kt = 0; kt < KT; ++kt) {
        __syncthreads();
        const int cur = kt & 1;
        if (kt + 1 < KT) load_tile(kt + 1);

        const __bf16* a = sA + cur * (BM * LDA);
        const __bf16* w = sW + cur * (BN * LDA);
#pragma unroll
        for (int ks = 0; ks < 4; ++ks) {
            const int kk = ks * 32;
            FragAB fa[4], fb[2];
#pragma unroll
            for (int i = 0; i < 4; ++i) {
                const __bf16* base = a + (wm * 64 + i * 16 + l16) * LDA + kk + lhalf * 8;
                fa[i].h[0] = *(const bf16x8*)(base);
                fa[i].h[1] = *(const bf16x8*)(base + 16);
            }
#pragma unroll
            for (int j = 0; j < 2; ++j) {
                const __bf16* base = w + (wn * 32 + j * 16 + l16) * LDA + kk + lhalf * 8;
                fb[j].h[0] = *(const bf16x8*)(base);
                fb[j].h[1] = *(const bf16x8*)(base + 16);
            }
#pragma unroll
            for (int i = 0; i < 4; ++i)
#pragma unroll
                for (int j = 0; j < 2; ++j)
                    acc[i][j] = __builtin_amdgcn_wmma_f32_16x16x32_bf16(
                        false, fa[i].v, false, fb[j].v, (short)0, acc[i][j], false, false);
        }
        if (kt + 1 < KT) store_tile((kt + 1) & 1);
    }

#pragma unroll
    for (int j = 0; j < 2; ++j) {
        const int n  = bn + wn * 32 + j * 16 + l16;
        const float b = bias[n];
#pragma unroll
        for (int i = 0; i < 4; ++i) {
            const int mrow = bm + wm * 64 + i * 16 + lhalf * 8;
#pragma unroll
            for (int r = 0; r < 8; ++r)
                __builtin_nontemporal_store(acc[i][j][r] + b,
                                            &out[(size_t)(mrow + r) * N + n]);
        }
    }
}

// =====================================================================
extern "C" void kernel_launch(void* const* d_in, const int* in_sizes, int n_in,
                              void* d_out, int out_size, void* d_ws, size_t ws_size,
                              hipStream_t stream) {
    const float* x       = (const float*)d_in[0];
    const int*   qw      = (const int*)  d_in[1];
    const float* wscales = (const float*)d_in[2];
    const float* wzeros  = (const float*)d_in[3];
    const float* bias    = (const float*)d_in[4];
    float*       out     = (float*)d_out;

    const int  N = in_sizes[4];                    // 12288
    const long K = (2L * in_sizes[1]) / N;         // 4096
    const long M = (long)in_sizes[0] / K;          // 4096

    const size_t xbytes = (size_t)M * K * sizeof(__bf16);
    const size_t wbytes = (size_t)N * K * sizeof(__bf16);

    if (ws_size >= xbytes + wbytes) {
        __bf16* xbf = (__bf16*)d_ws;
        __bf16* wbf = (__bf16*)((char*)d_ws + xbytes);

        const long xtot = M * K;
        convert_x_kernel<<<dim3((unsigned)((xtot / 8 + NTHREADS - 1) / NTHREADS)),
                           dim3(NTHREADS), 0, stream>>>(x, xbf, xtot);

        const long wtot = (long)N * (K / 8);       // int4 chunks
        dequant_w_kernel<<<dim3((unsigned)((wtot + NTHREADS - 1) / NTHREADS)),
                           dim3(NTHREADS), 0, stream>>>(qw, wscales, wzeros, wbf,
                                                        N, (int)K);

        dim3 grid((unsigned)(N / GBN), (unsigned)(M / GBM));
        const size_t smem_bytes = (size_t)2 * (GBM * GLDA + GBN * GLDA) * sizeof(__bf16);
        gemm_bf16_kernel<<<grid, dim3(NTHREADS), smem_bytes, stream>>>(
            xbf, wbf, bias, out, (int)M, N, (int)K);
    } else {
        dim3 grid((unsigned)(N / BN), (unsigned)(M / BM));
        const size_t smem_bytes = (size_t)2 * (BM * LDA + BN * LDA) * sizeof(__bf16);
        awq_w4a16_fused_kernel<<<grid, dim3(NTHREADS), smem_bytes, stream>>>(
            x, qw, wscales, wzeros, bias, out, (int)M, N, (int)K);
    }
}
